// MultiHeadAttention_60258391163205
// MI455X (gfx1250) — matrix operand order repaired and checked by reading, not compiled, not run on hardware
//
#include <hip/hip_runtime.h>

// ---------------------------------------------------------------------------
// MHA forward for gfx1250 (MI455X): bf16 WMMA GEMMs + shared-KV flash attention
// with double-buffered async global->LDS staging (ASYNCcnt partial waits).
// B=2, S=2048, E=1024, H=16, D=64.
// ---------------------------------------------------------------------------

#define B_   2
#define S_   2048
#define E_   1024
#define H_   16
#define D_   64
#define WSTRIDE 40   // ushort stride for 32-wide LDS rows (80B: 16B aligned)
#define KSTRIDE 72   // ushort stride for 64-wide LDS rows (144B: 16B aligned)

typedef __bf16          v16bf __attribute__((ext_vector_type(16)));
typedef float           v8f   __attribute__((ext_vector_type(8)));
typedef unsigned short  v16us __attribute__((ext_vector_type(16)));
typedef unsigned short  v8us  __attribute__((ext_vector_type(8)));
typedef unsigned short  v4us  __attribute__((ext_vector_type(4)));

// float -> bf16 raw bits via native hardware conversion
__device__ __forceinline__ unsigned short f2bf(float f) {
  return __builtin_bit_cast(unsigned short, (__bf16)f);
}

__device__ __forceinline__ v16bf make_frag(v8us lo, v8us hi) {
  v16us u;
#pragma unroll
  for (int i = 0; i < 8; ++i) { u[i] = lo[i]; u[i + 8] = hi[i]; }
  return __builtin_bit_cast(v16bf, u);
}

// 16 contiguous bf16 (32B, 16B-aligned) -> fragment
__device__ __forceinline__ v16bf load_frag16(const unsigned short* p) {
  v8us lo = *(const v8us*)p;
  v8us hi = *(const v8us*)(p + 8);
  return make_frag(lo, hi);
}

// two separated 8-element bf16 chunks -> fragment (A-operand pattern)
__device__ __forceinline__ v16bf load_frag8x2(const unsigned short* p0,
                                              const unsigned short* p1) {
  v8us lo = *(const v8us*)p0;
  v8us hi = *(const v8us*)p1;
  return make_frag(lo, hi);
}

__device__ __forceinline__ v8f wmma_bf16(v16bf a, v16bf b, v8f c) {
  return __builtin_amdgcn_wmma_f32_16x16x32_bf16(
      false, a, false, b, (short)0, c, false, false);
}

// ---- CDNA5 async global->LDS copy (ASYNCcnt-tracked, no VGPR round trip) ----
__device__ __forceinline__ void async_b128(unsigned lds_byte_off, const void* g) {
  asm volatile("global_load_async_to_lds_b128 %0, %1, off"
               :
               : "v"(lds_byte_off), "v"((unsigned long long)(size_t)g)
               : "memory");
}
// copy 32 contiguous bytes (16 bf16) global -> LDS
__device__ __forceinline__ void async_b256(unsigned lds_byte_off,
                                           const unsigned short* g) {
  async_b128(lds_byte_off, g);
  async_b128(lds_byte_off + 16, g + 8);
}
// partial waits: ASYNCcnt decrements in issue order, so "<= N" with N ops just
// issued guarantees the previous stage's N ops have landed in LDS.
#define WAIT_ASYNC(n) asm volatile("s_wait_asynccnt " #n ::: "memory")

__device__ __forceinline__ unsigned lds_off(const void* p) {
  return (unsigned)(size_t)p;   // LDS aperture: addr[31:0] == LDS byte offset
}

// ---------------------------------------------------------------------------
// Kernel 0a: convert x (f32) -> bf16, row-major [B*S, E].
// ---------------------------------------------------------------------------
__global__ __launch_bounds__(256) void convert_x_kernel(
    const float* __restrict__ x, unsigned short* __restrict__ xbf) {
  const size_t  i8 = ((size_t)blockIdx.x * 256 + threadIdx.x) * 8;
  const float4* p  = (const float4*)(x + i8);
  float4 a = p[0], b = p[1];
  v8us u;
  u[0] = f2bf(a.x); u[1] = f2bf(a.y); u[2] = f2bf(a.z); u[3] = f2bf(a.w);
  u[4] = f2bf(b.x); u[5] = f2bf(b.y); u[6] = f2bf(b.z); u[7] = f2bf(b.w);
  *(v8us*)(xbf + i8) = u;
}

// ---------------------------------------------------------------------------
// Kernel 0b: transpose+convert weights: Wt[n][k] = bf16(W[k][n]).
// 32x32 tiles through LDS; blockIdx.z selects which of the 4 weights.
// ---------------------------------------------------------------------------
__global__ __launch_bounds__(256) void wt_transpose_kernel(
    const float* __restrict__ W0, const float* __restrict__ W1,
    const float* __restrict__ W2, const float* __restrict__ W3,
    unsigned short* __restrict__ T0, unsigned short* __restrict__ T1,
    unsigned short* __restrict__ T2, unsigned short* __restrict__ T3) {
  __shared__ float tile[32][33];
  const float*    W;
  unsigned short* T;
  switch (blockIdx.z) {
    case 0:  W = W0; T = T0; break;
    case 1:  W = W1; T = T1; break;
    case 2:  W = W2; T = T2; break;
    default: W = W3; T = T3; break;
  }
  const int tid = threadIdx.x;
  const int nb  = blockIdx.x * 32;   // n tile
  const int kb  = blockIdx.y * 32;   // k tile

  {  // coalesced load: rows of W
    const int r  = tid >> 3;         // k within tile
    const int c4 = (tid & 7) * 4;    // n within tile
    float4 f = *(const float4*)(W + (size_t)(kb + r) * E_ + nb + c4);
    tile[r][c4 + 0] = f.x; tile[r][c4 + 1] = f.y;
    tile[r][c4 + 2] = f.z; tile[r][c4 + 3] = f.w;
  }
  __syncthreads();
  {  // coalesced store: rows of Wt (transposed)
    const int n  = tid >> 3;
    const int k4 = (tid & 7) * 4;
    v4us u;
#pragma unroll
    for (int j = 0; j < 4; ++j) u[j] = f2bf(tile[k4 + j][n]);
    *(v4us*)(T + (size_t)(nb + n) * E_ + kb + k4) = u;
  }
}

// ---------------------------------------------------------------------------
// Kernel 1: fused QKV projection, bf16 WMMA, double-buffered async staging.
// Block: 256 thr (8 waves); tile 16(M) x 128(N); grid (4096/16, 1024/128).
// Wt* are pre-transposed bf16 [N][K]. Outputs: Qbf,Kbf [B,H,S,D]; Vt [B,H,D,S].
// ---------------------------------------------------------------------------
__global__ __launch_bounds__(256) void qkv_proj_kernel(
    const unsigned short* __restrict__ xbf,
    const unsigned short* __restrict__ Wtq,
    const unsigned short* __restrict__ Wtk,
    const unsigned short* __restrict__ Wtv,
    const float* __restrict__ bq, const float* __restrict__ bk,
    const float* __restrict__ bv,
    unsigned short* __restrict__ Qbf, unsigned short* __restrict__ Kbf,
    unsigned short* __restrict__ Vt) {
  __shared__ unsigned short lds_w[2][3][128 * WSTRIDE];  // double buffered

  const int mb   = blockIdx.x;
  const int nb   = blockIdx.y;
  const int tid  = threadIdx.x;
  const int wave = tid >> 5;
  const int lane = tid & 31;
  const int hgrp = lane >> 4;
  const int lidx = lane & 15;

  v8f accq = {}, acck = {}, accv = {};

  const int             arow = mb * 16 + lidx;
  const unsigned short* xrow = xbf + (size_t)arow * E_;

  // staging assignment: each thread copies 32B of one Wt row per weight
  const int      nl   = tid >> 1;                      // 0..127
  const int      half = tid & 1;                       // 0/1
  const size_t   gsub = (size_t)(nb * 128 + nl) * E_ + half * 16;
  const unsigned so   = (unsigned)((nl * WSTRIDE + half * 16) * 2);
  const unsigned wsz  = 128 * WSTRIDE * 2;             // bytes per weight slab
  const unsigned lb[2] = {lds_off(&lds_w[0][0][0]) + so,
                          lds_off(&lds_w[1][0][0]) + so};

  // prologue: stage chunk 0 into buffer 0 (6 async b128 per thread)
  {
    const size_t g = gsub;
    async_b256(lb[0] + 0 * wsz, Wtq + g);
    async_b256(lb[0] + 1 * wsz, Wtk + g);
    async_b256(lb[0] + 2 * wsz, Wtv + g);
  }

  for (int kc = 0; kc < E_ / 32; ++kc) {
    const int cur = kc & 1;
    if (kc + 1 < E_ / 32) {   // stage next chunk into the other buffer
      const size_t   g  = gsub + (kc + 1) * 32;
      const unsigned l  = lb[1 - cur];
      async_b256(l + 0 * wsz, Wtq + g);
      async_b256(l + 1 * wsz, Wtk + g);
      async_b256(l + 2 * wsz, Wtv + g);
      WAIT_ASYNC(6);          // current chunk landed; next stays in flight
    } else {
      WAIT_ASYNC(0);
    }
    __syncthreads();

    const int   k0    = kc * 32 + hgrp * 16;
    const v16bf afrag = load_frag8x2(xrow + k0, xrow + k0 + 8);

    const int   boff = (wave * 16 + lidx) * WSTRIDE + hgrp * 16;
    const v16bf bfq  = load_frag16(&lds_w[cur][0][boff]);
    const v16bf bfk  = load_frag16(&lds_w[cur][1][boff]);
    const v16bf bfv  = load_frag16(&lds_w[cur][2][boff]);
    accq = wmma_bf16(afrag, bfq, accq);
    acck = wmma_bf16(afrag, bfk, acck);
    accv = wmma_bf16(afrag, bfv, accv);
    __syncthreads();   // all waves done reading 'cur' before it is re-staged
  }

  // Epilogue: scatter bf16 to Q/K ([B,H,S,D]) and Vt ([B,H,D,S]).
  const int   c   = nb * 128 + wave * 16 + lidx;
  const int   h   = c >> 6;
  const int   d   = c & 63;
  const float vbq = bq[c], vbk = bk[c], vbv = bv[c];
#pragma unroll
  for (int r = 0; r < 8; ++r) {
    const int m    = r + (hgrp << 3);
    const int srow = mb * 16 + m;
    const int b    = srow >> 11;
    const int s    = srow & (S_ - 1);
    const int bh   = b * H_ + h;
    Qbf[((size_t)bh * S_ + s) * D_ + d] = f2bf(accq[r] + vbq);
    Kbf[((size_t)bh * S_ + s) * D_ + d] = f2bf(acck[r] + vbk);
    Vt[((size_t)bh * D_ + d) * S_ + s]  = f2bf(accv[r] + vbv);
  }
}

// ---------------------------------------------------------------------------
// Kernel 2: causal flash attention, block-shared double-buffered K/V staging.
// Block: 128 thr (4 waves) share one (b,h); each wave owns a 16-query tile.
// ---------------------------------------------------------------------------
__global__ __launch_bounds__(128) void attn_kernel(
    const unsigned short* __restrict__ Qbf,
    const unsigned short* __restrict__ Kbf,
    const unsigned short* __restrict__ Vt,
    unsigned short* __restrict__ attn) {
  __shared__ unsigned short lds_k[2][32 * KSTRIDE];   // K tile: 32 keys x 64 d
  __shared__ unsigned short lds_v[2][64 * WSTRIDE];   // V tile: 64 d x 32 keys
  __shared__ unsigned short lds_p[4][16 * WSTRIDE];   // per-wave P tile 16x32

  const int tid  = threadIdx.x;
  const int wave = tid >> 5;
  const int lane = tid & 31;
  const int hgrp = lane >> 4;
  const int lidx = lane & 15;

  const int qb64   = blockIdx.x & 31;
  const int bh     = blockIdx.x >> 5;
  const int q_base = qb64 * 64 + wave * 16;

  const unsigned short* qp = Qbf + (size_t)bh * S_ * D_;
  const unsigned short* kp = Kbf + (size_t)bh * S_ * D_;
  const unsigned short* vp = Vt + (size_t)bh * D_ * S_;

  // Q A-fragments for the two 32-wide D chunks, resident all loop
  const int sq = q_base + lidx;
  v16bf     qfrag[2];
#pragma unroll
  for (int c = 0; c < 2; ++c) {
    const int k0 = c * 32 + hgrp * 16;
    qfrag[c] = load_frag8x2(qp + (size_t)sq * D_ + k0,
                            qp + (size_t)sq * D_ + k0 + 8);
  }

  v8f   O[4] = {{}, {}, {}, {}};
  float mrow[8], lrow[8];
#pragma unroll
  for (int r = 0; r < 8; ++r) { mrow[r] = -3.0e38f; lrow[r] = 0.0f; }

  const float scale   = 0.03125f;                    // 1/sqrt(E) = 1/32
  const int   nkt     = (q_base + 16 + 31) >> 5;     // this wave's chunks
  const int   nkt_max = qb64 * 2 + 2;                // last wave's chunks

  // staging assignment (whole block): K row = tid>>2, 32B quarter = tid&3;
  //                                   V row d = tid>>1, 32B half = tid&1.
  const int      krow = tid >> 2, kq = tid & 3;
  const int      vrow = tid >> 1, vh = tid & 1;
  const unsigned lko[2] = {
      lds_off(lds_k[0]) + (unsigned)((krow * KSTRIDE + kq * 16) * 2),
      lds_off(lds_k[1]) + (unsigned)((krow * KSTRIDE + kq * 16) * 2)};
  const unsigned lvo[2] = {
      lds_off(lds_v[0]) + (unsigned)((vrow * WSTRIDE + vh * 16) * 2),
      lds_off(lds_v[1]) + (unsigned)((vrow * WSTRIDE + vh * 16) * 2)};

  // prologue: stage chunk 0 into buffer 0 (4 async b128 per thread)
  async_b256(lko[0], kp + (size_t)krow * D_ + kq * 16);
  async_b256(lvo[0], vp + (size_t)vrow * S_ + vh * 16);

  for (int t = 0; t < nkt_max; ++t) {
    const int kb  = t * 32;
    const int cur = t & 1;
    if (t + 1 < nkt_max) {
      const int nkb = kb + 32;
      async_b256(lko[1 - cur], kp + (size_t)(nkb + krow) * D_ + kq * 16);
      async_b256(lvo[1 - cur], vp + (size_t)vrow * S_ + nkb + vh * 16);
      WAIT_ASYNC(4);
    } else {
      WAIT_ASYNC(0);
    }
    __syncthreads();

    if (t < nkt) {
      // ---- scores: two 16x16 accumulators over the 32-key chunk ----
      const v16bf bk00 = load_frag16(&lds_k[cur][lidx * KSTRIDE + hgrp * 16]);
      const v16bf bk01 = load_frag16(&lds_k[cur][lidx * KSTRIDE + 32 + hgrp * 16]);
      const v16bf bk10 = load_frag16(&lds_k[cur][(16 + lidx) * KSTRIDE + hgrp * 16]);
      const v16bf bk11 = load_frag16(&lds_k[cur][(16 + lidx) * KSTRIDE + 32 + hgrp * 16]);
      v8f sc0 = {}, sc1 = {};
      sc0 = wmma_bf16(qfrag[0], bk00, sc0);
      sc0 = wmma_bf16(qfrag[1], bk01, sc0);
      sc1 = wmma_bf16(qfrag[0], bk10, sc1);
      sc1 = wmma_bf16(qfrag[1], bk11, sc1);

      const bool need_mask = (kb + 31) > q_base;
#pragma unroll
      for (int r = 0; r < 8; ++r) {
        sc0[r] *= scale;
        sc1[r] *= scale;
        const int m    = r + (hgrp << 3);
        const int qrow = q_base + m;
        if (need_mask) {
          if (kb + lidx > qrow)      sc0[r] = -3.0e38f;
          if (kb + 16 + lidx > qrow) sc1[r] = -3.0e38f;
        }
        float rm = fmaxf(sc0[r], sc1[r]);
        rm = fmaxf(rm, __shfl_xor(rm, 1, 32));
        rm = fmaxf(rm, __shfl_xor(rm, 2, 32));
        rm = fmaxf(rm, __shfl_xor(rm, 4, 32));
        rm = fmaxf(rm, __shfl_xor(rm, 8, 32));

        const float mnew = fmaxf(mrow[r], rm);
        const float corr = __expf(mrow[r] - mnew);
        sc0[r] = __expf(sc0[r] - mnew);
        sc1[r] = __expf(sc1[r] - mnew);

        float rs = sc0[r] + sc1[r];
        rs += __shfl_xor(rs, 1, 32);
        rs += __shfl_xor(rs, 2, 32);
        rs += __shfl_xor(rs, 4, 32);
        rs += __shfl_xor(rs, 8, 32);

        lrow[r] = lrow[r] * corr + rs;
        mrow[r] = mnew;
        O[0][r] *= corr;
        O[1][r] *= corr;
        O[2][r] *= corr;
        O[3][r] *= corr;

        lds_p[wave][m * WSTRIDE + lidx]      = f2bf(sc0[r]);
        lds_p[wave][m * WSTRIDE + 16 + lidx] = f2bf(sc1[r]);
      }

      // ---- P*V: P re-read in A layout; V tile already B-layout friendly ----
      const unsigned short* pp    = &lds_p[wave][lidx * WSTRIDE + hgrp * 16];
      const v16bf           pfrag = load_frag8x2(pp, pp + 8);
      const v16bf bv0 = load_frag16(&lds_v[cur][(0 * 16 + lidx) * WSTRIDE + hgrp * 16]);
      const v16bf bv1 = load_frag16(&lds_v[cur][(1 * 16 + lidx) * WSTRIDE + hgrp * 16]);
      const v16bf bv2 = load_frag16(&lds_v[cur][(2 * 16 + lidx) * WSTRIDE + hgrp * 16]);
      const v16bf bv3 = load_frag16(&lds_v[cur][(3 * 16 + lidx) * WSTRIDE + hgrp * 16]);
      O[0] = wmma_bf16(pfrag, bv0, O[0]);
      O[1] = wmma_bf16(pfrag, bv1, O[1]);
      O[2] = wmma_bf16(pfrag, bv2, O[2]);
      O[3] = wmma_bf16(pfrag, bv3, O[3]);
    }
    __syncthreads();   // all waves done with 'cur' before it is re-staged
  }

  // ---- normalize + store attn output bf16 row-major [B,S,E] ----
  const int b_ = bh >> 4;
  const int h_ = bh & 15;
#pragma unroll
  for (int r = 0; r < 8; ++r) {
    const float inv = 1.0f / lrow[r];
    const int   m   = r + (hgrp << 3);
    const int   s   = q_base + m;
    unsigned short* orow = attn + ((size_t)(b_ * S_ + s)) * E_ + h_ * 64 + lidx;
#pragma unroll
    for (int dc = 0; dc < 4; ++dc) orow[dc * 16] = f2bf(O[dc][r] * inv);
  }
}

// ---------------------------------------------------------------------------
// Kernel 3: output projection  out = attn(bf16) @ Wo + bo  (f32 out).
// ---------------------------------------------------------------------------
__global__ __launch_bounds__(256) void out_proj_kernel(
    const unsigned short* __restrict__ attn,
    const unsigned short* __restrict__ Wto,
    const float* __restrict__ bo, float* __restrict__ out) {
  __shared__ unsigned short lds_w[2][128 * WSTRIDE];

  const int mb   = blockIdx.x;
  const int nb   = blockIdx.y;
  const int tid  = threadIdx.x;
  const int wave = tid >> 5;
  const int lane = tid & 31;
  const int hgrp = lane >> 4;
  const int lidx = lane & 15;

  v8f acc = {};

  const int             arow = mb * 16 + lidx;
  const unsigned short* ap   = attn + (size_t)arow * E_;

  const int      nl   = tid >> 1;
  const int      half = tid & 1;
  const size_t   gsub = (size_t)(nb * 128 + nl) * E_ + half * 16;
  const unsigned so   = (unsigned)((nl * WSTRIDE + half * 16) * 2);
  const unsigned lw[2] = {lds_off(lds_w[0]) + so, lds_off(lds_w[1]) + so};

  async_b256(lw[0], Wto + gsub);   // prologue: stage chunk 0

  for (int kc = 0; kc < E_ / 32; ++kc) {
    const int cur = kc & 1;
    if (kc + 1 < E_ / 32) {
      async_b256(lw[1 - cur], Wto + gsub + (kc + 1) * 32);
      WAIT_ASYNC(2);
    } else {
      WAIT_ASYNC(0);
    }
    __syncthreads();

    const int   k0    = kc * 32 + hgrp * 16;
    const v16bf afrag = load_frag8x2(ap + k0, ap + k0 + 8);
    const v16bf bfrag =
        load_frag16(&lds_w[cur][(wave * 16 + lidx) * WSTRIDE + hgrp * 16]);
    acc = wmma_bf16(afrag, bfrag, acc);
    __syncthreads();
  }

  const int   c    = nb * 128 + wave * 16 + lidx;
  const float bias = bo[c];
#pragma unroll
  for (int r = 0; r < 8; ++r) {
    const int m    = r + (hgrp << 3);
    const int srow = mb * 16 + m;
    out[(size_t)srow * E_ + c] = acc[r] + bias;
  }
}

// ---------------------------------------------------------------------------
extern "C" void kernel_launch(void* const* d_in, const int* in_sizes, int n_in,
                              void* d_out, int out_size, void* d_ws,
                              size_t ws_size, hipStream_t stream) {
  (void)in_sizes; (void)n_in; (void)out_size; (void)ws_size;
  const float* x  = (const float*)d_in[0];
  const float* Wq = (const float*)d_in[1];
  const float* bq = (const float*)d_in[2];
  const float* Wk = (const float*)d_in[3];
  const float* bk = (const float*)d_in[4];
  const float* Wv = (const float*)d_in[5];
  const float* bv = (const float*)d_in[6];
  const float* Wo = (const float*)d_in[7];
  const float* bo = (const float*)d_in[8];
  float*       out = (float*)d_out;

  // workspace layout (ushort elements): 48MB total
  const size_t    NQ = (size_t)B_ * S_ * E_;   // 4M (x / per-tensor Q,K,V,attn)
  const size_t    NW = (size_t)E_ * E_;        // 1M per weight
  unsigned short* Xbf  = (unsigned short*)d_ws;
  unsigned short* Wtq  = Xbf + NQ;
  unsigned short* Wtk  = Wtq + NW;
  unsigned short* Wtv  = Wtk + NW;
  unsigned short* Wto  = Wtv + NW;
  unsigned short* Qbf  = Wto + NW;
  unsigned short* Kbf  = Qbf + NQ;
  unsigned short* Vt   = Kbf + NQ;
  unsigned short* attn = Vt + NQ;

  convert_x_kernel<<<dim3((unsigned)(NQ / 2048)), 256, 0, stream>>>(x, Xbf);
  wt_transpose_kernel<<<dim3(E_ / 32, E_ / 32, 4), 256, 0, stream>>>(
      Wq, Wk, Wv, Wo, Wtq, Wtk, Wtv, Wto);

  qkv_proj_kernel<<<dim3((B_ * S_) / 16, E_ / 128), 256, 0, stream>>>(
      Xbf, Wtq, Wtk, Wtv, bq, bk, bv, Qbf, Kbf, Vt);

  attn_kernel<<<dim3(B_ * H_ * (S_ / 64)), 128, 0, stream>>>(Qbf, Kbf, Vt, attn);

  out_proj_kernel<<<dim3((B_ * S_) / 16, E_ / 128), 256, 0, stream>>>(
      attn, Wto, bo, out);
}
